// FlatTorusPhasorBlock_37082747633868
// MI455X (gfx1250) — matrix operand order repaired and checked by reading, not compiled, not run on hardware
//
#include <hip/hip_runtime.h>
#include <hip/hip_bf16.h>
#include <math.h>

// ---------------------------------------------------------------------------
// FlatTorusPhasorBlock == causal linear attention with cos/sin feature maps:
//   retrieved[l,d] = sum_{l'<=l} (cos_q[l].cos_k[l'] + sin_q[l].sin_k[l']) V[l',d]
// All GEMMs run on v_wmma_f32_16x16x32_f16 (f16 in, f32 accum).
// ---------------------------------------------------------------------------

typedef _Float16 half_t;
typedef _Float16 v16h __attribute__((ext_vector_type(16)));
typedef _Float16 v8h  __attribute__((ext_vector_type(8)));
typedef float    v8f  __attribute__((ext_vector_type(8)));

#define Dm 256
#define Km 64
#define Bm 2
#define Lm 1024
#define NROWS (Bm * Lm)   // 2048

__device__ __forceinline__ v16h pack16(v8h lo, v8h hi) {
  v16h r;
#pragma unroll
  for (int t = 0; t < 8; ++t) { r[t] = lo[t]; r[t + 8] = hi[t]; }
  return r;
}

// A fragment (16x32, MxK) from row-major A. ISA layout (CDNA5 ISA):
// lane<16: M=lane,   halves 0..7 = K k0..k0+7,   8..15 = K k0+16..k0+23
// lane>=16:M=lane-16,halves 0..7 = K k0+8..k0+15,8..15 = K k0+24..k0+31
__device__ __forceinline__ v16h load_a_frag(const half_t* A, int lda, int row0,
                                            int k0, int lane) {
  int r  = row0 + (lane & 15);
  int kb = k0 + ((lane >> 4) << 3);
  const half_t* p = A + (size_t)r * lda + kb;
  return pack16(*(const v8h*)p, *(const v8h*)(p + 16));
}

// B fragment (32x16, KxN) from TRANSPOSED (NxK row-major) storage:
// lane<16: N=col0+lane,    halves 0..7 = K k0..k0+7,   8..15 = K k0+16..k0+23
// lane>=16:N=col0+lane-16, halves 0..7 = K k0+8..k0+15,8..15 = K k0+24..k0+31
__device__ __forceinline__ v16h load_b_frag(const half_t* Bt, int ldb, int col0,
                                            int k0, int lane) {
  int n  = col0 + (lane & 15);
  int kb = k0 + ((lane >> 4) << 3);
  const half_t* p = Bt + (size_t)n * ldb + kb;
  return pack16(*(const v8h*)p, *(const v8h*)(p + 16));
}

__device__ __forceinline__ v8f wmma_f16(v16h a, v16h b, v8f c) {
  return __builtin_amdgcn_wmma_f32_16x16x32_f16(false, a, false, b, (short)0, c,
                                                false, false);
}

template <int KDIM>
__device__ __forceinline__ v8f gemm_tile(const half_t* A, int lda,
                                         const half_t* Bt, int ldb, int row0,
                                         int col0, int lane) {
  v8f c = {};
#pragma unroll
  for (int k = 0; k < KDIM; k += 32) {
    v16h a = load_a_frag(A, lda, row0, k, lane);
    v16h b = load_b_frag(Bt, ldb, col0, k, lane);
    c = wmma_f16(a, b, c);
  }
  return c;
}

// ------------------------- prep kernels -----------------------------------
__global__ void k_cvt_f16(const float* __restrict__ src, half_t* __restrict__ dst,
                          int n) {
  int i = blockIdx.x * blockDim.x + threadIdx.x;
  if (i < n) dst[i] = (half_t)src[i];
}

// w: (fin x fout) row-major f32  ->  wt: (fout x fin) row-major f16
__global__ void k_transpose_f16(const float* __restrict__ w,
                                half_t* __restrict__ wt, int fin, int fout) {
  int i = blockIdx.x * blockDim.x + threadIdx.x;
  if (i < fin * fout) {
    int o = i / fin, in = i % fin;
    wt[i] = (half_t)w[(size_t)in * fout + o];
  }
}

// ------------------------- MLP layer 1: gelu(x@W1 + b1) -------------------
__global__ void k_mlp1(const half_t* __restrict__ X, const half_t* __restrict__ W1t,
                       const float* __restrict__ b1, half_t* __restrict__ H) {
  int lane = threadIdx.x & 31;
  int tile = blockIdx.x * (blockDim.x >> 5) + (threadIdx.x >> 5);
  int row0 = (tile >> 4) * 16, col0 = (tile & 15) * 16;  // 128 x 16 tiles
  v8f c = gemm_tile<Dm>(X, Dm, W1t, Dm, row0, col0, lane);
  int col = col0 + (lane & 15);
  float bias = b1[col];
  int rsel = (lane >> 4) << 3;
#pragma unroll
  for (int v = 0; v < 8; ++v) {
    int row = row0 + v + rsel;
    float h = c[v] + bias;
    float g = 0.5f * h * (1.0f + erff(h * 0.70710678118654752f));  // exact gelu
    H[(size_t)row * Dm + col] = (half_t)g;
  }
}

// ---------------- MLP layer 2: phase = pi*tanh(H@W2+b2) -> cos/sin feats --
__global__ void k_mlp2(const half_t* __restrict__ H, const half_t* __restrict__ W2t,
                       const float* __restrict__ b2, half_t* __restrict__ F) {
  int lane = threadIdx.x & 31;
  int tile = blockIdx.x * (blockDim.x >> 5) + (threadIdx.x >> 5);
  int row0 = (tile >> 2) * 16, col0 = (tile & 3) * 16;  // 128 x 4 tiles
  v8f c = gemm_tile<Dm>(H, Dm, W2t, Dm, row0, col0, lane);
  int col = col0 + (lane & 15);
  float bias = b2[col];
  int rsel = (lane >> 4) << 3;
#pragma unroll
  for (int v = 0; v < 8; ++v) {
    int row = row0 + v + rsel;
    float p = tanhf(c[v] + bias) * 3.14159265358979323846f;
    F[(size_t)row * (2 * Km) + col]      = (half_t)cosf(p);
    F[(size_t)row * (2 * Km) + Km + col] = (half_t)sinf(p);
  }
}

// ---------------- V projection, written directly transposed (B,D,L) -------
__global__ void k_vproj(const half_t* __restrict__ X, const half_t* __restrict__ Vwt,
                        const float* __restrict__ vb, half_t* __restrict__ Vt) {
  int lane = threadIdx.x & 31;
  int tile = blockIdx.x * (blockDim.x >> 5) + (threadIdx.x >> 5);
  int row0 = (tile >> 4) * 16, col0 = (tile & 15) * 16;
  v8f c = gemm_tile<Dm>(X, Dm, Vwt, Dm, row0, col0, lane);
  int col = col0 + (lane & 15);
  float bias = vb[col];
  int rsel = (lane >> 4) << 3;
#pragma unroll
  for (int v = 0; v < 8; ++v) {
    int row = row0 + v + rsel;
    int b = row >> 10, l = row & (Lm - 1);
    Vt[((size_t)b * Dm + col) * Lm + l] = (half_t)(c[v] + bias);
  }
}

// ---------------- fused causal linear attention ---------------------------
// grid: B * L/16 blocks; 8 waves/block, wave w owns output cols [32w,32w+32)
__global__ void k_attn(const half_t* __restrict__ Qf, const half_t* __restrict__ Kf,
                       const half_t* __restrict__ Vt, float* __restrict__ Rv) {
  __shared__ half_t sS[8][16 * 32];  // per-wave S-tile staging (C-frag -> A-frag)
  int lane = threadIdx.x & 31;
  int wave = threadIdx.x >> 5;
  int b = blockIdx.x >> 6;
  int i = blockIdx.x & 63;  // query tile
  const half_t* Qb = Qf + (size_t)b * Lm * (2 * Km);
  const half_t* Kb = Kf + (size_t)b * Lm * (2 * Km);
  const half_t* Vb = Vt + (size_t)b * Dm * Lm;

  v16h qa[4];
#pragma unroll
  for (int t = 0; t < 4; ++t) qa[t] = load_a_frag(Qb, 2 * Km, i * 16, 32 * t, lane);

  v8f acc0 = {}, acc1 = {};
  half_t* my = &sS[wave][0];
  int colN = lane & 15;
  int rsel = (lane >> 4) << 3;

  for (int j2 = 0; j2 <= i; j2 += 2) {
#pragma unroll
    for (int h = 0; h < 2; ++h) {
      int j = j2 + h;          // key tile (wave-uniform)
      v8f s = {};
      if (j <= i) {            // uniform branch: EXEC stays full around WMMA
#pragma unroll
        for (int t = 0; t < 4; ++t) {
          v16h kb = load_b_frag(Kb, 2 * Km, j * 16, 32 * t, lane);
          s = wmma_f16(qa[t], kb, s);
        }
      }
      bool diag = (j == i);
#pragma unroll
      for (int v = 0; v < 8; ++v) {
        int row = v + rsel;
        float sv = s[v];
        if (j > i || (diag && colN > row)) sv = 0.0f;  // causal mask
        my[row * 32 + h * 16 + colN] = (half_t)sv;
      }
    }
    asm volatile("" ::: "memory");  // keep LDS stores before the re-load
    v16h sa  = load_a_frag(my, 32, 0, 0, lane);          // S chunk as A (16x32)
    v16h vb0 = load_b_frag(Vb, Lm, wave * 32,      j2 * 16, lane);
    v16h vb1 = load_b_frag(Vb, Lm, wave * 32 + 16, j2 * 16, lane);
    acc0 = wmma_f16(sa, vb0, acc0);
    acc1 = wmma_f16(sa, vb1, acc1);
  }

#pragma unroll
  for (int v = 0; v < 8; ++v) {
    int row = v + rsel;
    int l = i * 16 + row;
    float scale = rsqrtf((float)(l + 1) * (float)Km);
    size_t base = ((size_t)(b * Lm + l)) * Dm + wave * 32;
    Rv[base + colN]      = acc0[v] * scale;
    Rv[base + 16 + colN] = acc1[v] * scale;
  }
}

// ---------------- layernorm over D ----------------------------------------
__global__ void k_ln(const float* __restrict__ Rv, const float* __restrict__ g,
                     const float* __restrict__ bb, half_t* __restrict__ Rln) {
  __shared__ float red[Dm];
  int row = blockIdx.x, t = threadIdx.x;
  float v = Rv[(size_t)row * Dm + t];
  red[t] = v;
  __syncthreads();
  for (int s = Dm / 2; s > 0; s >>= 1) {
    if (t < s) red[t] += red[t + s];
    __syncthreads();
  }
  float mean = red[0] * (1.0f / Dm);
  __syncthreads();
  float d = v - mean;
  red[t] = d * d;
  __syncthreads();
  for (int s = Dm / 2; s > 0; s >>= 1) {
    if (t < s) red[t] += red[t + s];
    __syncthreads();
  }
  float var = red[0] * (1.0f / Dm);
  Rln[(size_t)row * Dm + t] = (half_t)(g[t] * d * rsqrtf(var + 1e-5f) + bb[t]);
}

// ---------------- output projection + residual ----------------------------
__global__ void k_out(const half_t* __restrict__ Rln, const half_t* __restrict__ Owt,
                      const float* __restrict__ ob, const float* __restrict__ X,
                      float* __restrict__ Out) {
  int lane = threadIdx.x & 31;
  int tile = blockIdx.x * (blockDim.x >> 5) + (threadIdx.x >> 5);
  int row0 = (tile >> 4) * 16, col0 = (tile & 15) * 16;
  v8f c = gemm_tile<Dm>(Rln, Dm, Owt, Dm, row0, col0, lane);
  int col = col0 + (lane & 15);
  float bias = ob[col];
  int rsel = (lane >> 4) << 3;
#pragma unroll
  for (int v = 0; v < 8; ++v) {
    int row = row0 + v + rsel;
    size_t idx = (size_t)row * Dm + col;
    Out[idx] = X[idx] + c[v] + bias;
  }
}

// ---------------------------------------------------------------------------
extern "C" void kernel_launch(void* const* d_in, const int* in_sizes, int n_in,
                              void* d_out, int out_size, void* d_ws, size_t ws_size,
                              hipStream_t stream) {
  const float* x   = (const float*)d_in[0];
  const float* kw1 = (const float*)d_in[1];
  const float* kb1 = (const float*)d_in[2];
  const float* kw2 = (const float*)d_in[3];
  const float* kb2 = (const float*)d_in[4];
  const float* qw1 = (const float*)d_in[5];
  const float* qb1 = (const float*)d_in[6];
  const float* qw2 = (const float*)d_in[7];
  const float* qb2 = (const float*)d_in[8];
  const float* vw  = (const float*)d_in[9];
  const float* vb  = (const float*)d_in[10];
  const float* lng = (const float*)d_in[11];
  const float* lnb = (const float*)d_in[12];
  const float* ow  = (const float*)d_in[13];
  const float* ob  = (const float*)d_in[14];
  float* out = (float*)d_out;

  char* ws = (char*)d_ws;
  size_t off = 0;
  auto alloc = [&](size_t bytes) -> char* {
    char* p = ws + off;
    off = (off + bytes + 255) & ~(size_t)255;
    return p;
  };
  half_t* Xh   = (half_t*)alloc((size_t)NROWS * Dm * 2);
  half_t* KW1t = (half_t*)alloc((size_t)Dm * Dm * 2);
  half_t* QW1t = (half_t*)alloc((size_t)Dm * Dm * 2);
  half_t* VWt  = (half_t*)alloc((size_t)Dm * Dm * 2);
  half_t* OWt  = (half_t*)alloc((size_t)Dm * Dm * 2);
  half_t* KW2t = (half_t*)alloc((size_t)Km * Dm * 2);
  half_t* QW2t = (half_t*)alloc((size_t)Km * Dm * 2);
  half_t* Hk   = (half_t*)alloc((size_t)NROWS * Dm * 2);
  half_t* Hq   = (half_t*)alloc((size_t)NROWS * Dm * 2);
  half_t* Kf   = (half_t*)alloc((size_t)NROWS * 2 * Km * 2);
  half_t* Qf   = (half_t*)alloc((size_t)NROWS * 2 * Km * 2);
  half_t* Vt   = (half_t*)alloc((size_t)Bm * Dm * Lm * 2);
  float*  Rv   = (float*)alloc((size_t)NROWS * Dm * 4);
  half_t* Rln  = (half_t*)alloc((size_t)NROWS * Dm * 2);

  // prep: f16 x, transposed f16 weights
  k_cvt_f16<<<dim3((NROWS * Dm + 255) / 256), dim3(256), 0, stream>>>(x, Xh, NROWS * Dm);
  k_transpose_f16<<<dim3(Dm * Dm / 256), dim3(256), 0, stream>>>(kw1, KW1t, Dm, Dm);
  k_transpose_f16<<<dim3(Dm * Dm / 256), dim3(256), 0, stream>>>(qw1, QW1t, Dm, Dm);
  k_transpose_f16<<<dim3(Dm * Dm / 256), dim3(256), 0, stream>>>(vw, VWt, Dm, Dm);
  k_transpose_f16<<<dim3(Dm * Dm / 256), dim3(256), 0, stream>>>(ow, OWt, Dm, Dm);
  k_transpose_f16<<<dim3(Dm * Km / 256), dim3(256), 0, stream>>>(kw2, KW2t, Dm, Km);
  k_transpose_f16<<<dim3(Dm * Km / 256), dim3(256), 0, stream>>>(qw2, QW2t, Dm, Km);

  // phase MLPs -> cos/sin features; V projection (stored transposed)
  k_mlp1<<<dim3(512), dim3(128), 0, stream>>>(Xh, KW1t, kb1, Hk);
  k_mlp1<<<dim3(512), dim3(128), 0, stream>>>(Xh, QW1t, qb1, Hq);
  k_mlp2<<<dim3(128), dim3(128), 0, stream>>>(Hk, KW2t, kb2, Kf);
  k_mlp2<<<dim3(128), dim3(128), 0, stream>>>(Hq, QW2t, qb2, Qf);
  k_vproj<<<dim3(512), dim3(128), 0, stream>>>(Xh, VWt, vb, Vt);

  // fused causal linear attention + position scaling
  k_attn<<<dim3(Bm * Lm / 16), dim3(256), 0, stream>>>(Qf, Kf, Vt, Rv);

  // layernorm, output projection, residual
  k_ln<<<dim3(NROWS), dim3(Dm), 0, stream>>>(Rv, lng, lnb, Rln);
  k_out<<<dim3(512), dim3(128), 0, stream>>>(Rln, OWt, ob, x, out);
}
